// DenseEnergyLoss_13005160972869
// MI455X (gfx1250) — matrix operand order repaired and checked by reading, not compiled, not run on hardware
//
#include <hip/hip_runtime.h>

// ---------------------------------------------------------------------------
// DenseEnergyLoss (fused, flash-attention style) for MI455X / gfx1250
//   loss = -(1e-7/4) * sum_n <W_n, M_n>_F
//   W[i,j] = exp(-0.5*max(sq_i+sq_j-2*f_i.f_j,0))  (5-D bilateral features)
//   M[i,j] = sum_k (gate_i*segroi_k[i])*(segroi_k[j])  (K=21)
// Never materializes the 64MB/image W matrix; all operands L2-resident.
// G via v_wmma_f32_16x16x4_f32 (x2, K=8), M via v_wmma_f32_16x16x32_f16.
// ---------------------------------------------------------------------------

typedef __attribute__((ext_vector_type(16))) _Float16 v16h;
typedef __attribute__((ext_vector_type(8)))  float    v8f;
typedef __attribute__((ext_vector_type(2)))  float    v2f;

#define NIMG  4
#define KCH   21
#define HIN   128
#define HO    64
#define PPIX  4096          // HO*HO
#define NTILE 256           // PPIX/16

// workspace layout, per image (bytes)
#define OFF_F    0u         // PPIX * 8 f32  (features, 5 used + 3 zero pad)
#define OFF_SQ   131072u    // PPIX f32      (|f|^2)
#define OFF_SA   147456u    // NTILE*512 f16 (gate*segroi, WMMA A layout, K pad 32)
#define OFF_SB   409600u    // NTILE*512 f16 (segroi,      WMMA B layout)
#define IMG_STRIDE 671744u  // total 4 * 671744 = ~2.6 MB of d_ws

// ---------------------------------------------------------------------------
// Prep: downsample (nearest = stride-2, bilinear = 2x2 avg), gate, and pack
// seg tiles directly into the gfx1250 WMMA f16 A/B register layouts so the
// main kernel loads each operand as one contiguous 32B chunk per lane.
// One thread per output pixel (n,p).
// ---------------------------------------------------------------------------
__global__ void dense_energy_prep(const float* __restrict__ images,
                                  const float* __restrict__ segs,
                                  const float* __restrict__ rois,
                                  const int*   __restrict__ labels,
                                  float* __restrict__ out,
                                  unsigned char* __restrict__ ws)
{
    int idx = blockIdx.x * blockDim.x + threadIdx.x;   // 0 .. NIMG*PPIX-1
    if (idx == 0) out[0] = 0.0f;                       // zero the accumulator
    int n = idx >> 12;
    int p = idx & (PPIX - 1);
    int y = p >> 6, x = p & 63;
    int iy = 2 * y, ix = 2 * x;

    unsigned char* base = ws + (unsigned)n * IMG_STRIDE;
    float*    F  = (float*)   (base + OFF_F);
    float*    SQ = (float*)   (base + OFF_SQ);
    _Float16* SA = (_Float16*)(base + OFF_SA);
    _Float16* SB = (_Float16*)(base + OFF_SB);

    // ---- bilateral features: [x/50, y/50, rgb/15], sigma_xy*SCALE = 50
    float f[8];
    f[0] = (float)x * (1.0f / 50.0f);
    f[1] = (float)y * (1.0f / 50.0f);
    #pragma unroll
    for (int c = 0; c < 3; ++c)
        f[2 + c] = images[((n * 3 + c) * HIN + iy) * HIN + ix] * (1.0f / 15.0f);
    f[5] = f[6] = f[7] = 0.0f;
    float sq = 0.0f;
    #pragma unroll
    for (int i = 0; i < 5; ++i) sq += f[i] * f[i];
    #pragma unroll
    for (int i = 0; i < 8; ++i) F[p * 8 + i] = f[i];
    SQ[p] = sq;

    // ---- seg bilinear x0.5 == exact 2x2 average pool; gate from ROI/labels
    float sgv[KCH];
    float mx = -1e30f;
    #pragma unroll
    for (int k = 0; k < KCH; ++k) {
        const float* s = segs + ((n * KCH + k) * HIN + iy) * HIN + ix;
        float v = 0.25f * (s[0] + s[1] + s[HIN] + s[HIN + 1]);
        sgv[k] = v;
        mx = fmaxf(mx, v);
    }
    float roi  = rois[(n * HIN + iy) * HIN + ix];
    int   lbl  = labels[(n * HIN + iy) * HIN + ix];
    float gate = (lbl == 255) ? 1.0f : (roi - mx);
    gate = fmaxf(gate, 0.0f);

    // ---- pack into WMMA tile layouts (CDNA5 ISA 7.12.2, 16-bit, wave32)
    int t = p >> 4;      // tile
    int m = p & 15;      // row (A) / col (B) inside tile
    #pragma unroll
    for (int k = 0; k < 32; ++k) {
        float b = (k < KCH) ? sgv[k] * roi : 0.0f;   // segroi
        float a = gate * b;                           // gate*segroi
        // A-matrix 16x32: lanes0-15 K{0-7,16-23}, lanes16-31 K{8-15,24-31}
        int grpA  = ((k >= 8) && (k < 16)) || (k >= 24);
        int halfA = (k < 8) ? k : (k < 16 ? k - 8 : (k < 24 ? k - 8 : k - 16));
        SA[t * 512 + (m + (grpA ? 16 : 0)) * 16 + halfA] = (_Float16)a;
        // B-matrix 32x16: lanes0-15 K{0-7,16-23} (col N=lane), lanes16-31 K{8-15,24-31}
        SB[t * 512 + (m + (((k >> 3) & 1) ? 16 : 0)) * 16 + ((k & 7) + 8 * (k >> 4))] = (_Float16)b;
    }
}

// ---------------------------------------------------------------------------
// Main: grid (256 i-tiles x 4 images), 8 waves/block; wave w sweeps j-tiles
// w, w+8, ... Per tile pair: 2x v_wmma_f32_16x16x4_f32 (Gram, K=8) +
// 1x v_wmma_f32_16x16x32_f16 (channel contraction, K=32) + exp/fma epilogue.
// ---------------------------------------------------------------------------
__global__ void dense_energy_main(const unsigned char* __restrict__ ws,
                                  float* __restrict__ out)
{
    int itile = blockIdx.x;
    int n     = blockIdx.y;
    int wave  = threadIdx.x >> 5;
    int lane  = threadIdx.x & 31;
    int m     = lane & 15;
    int hi    = lane >> 4;

    const unsigned char* base = ws + (unsigned)n * IMG_STRIDE;
    const float*    F  = (const float*)   (base + OFF_F);
    const float*    SQ = (const float*)   (base + OFF_SQ);
    const _Float16* SA = (const _Float16*)(base + OFF_SA);
    const _Float16* SB = (const _Float16*)(base + OFF_SB);

    // loop-invariant i-side operands
    v16h sa  = *(const v16h*)(SA + itile * 512 + lane * 16);          // 32B/lane
    v8f  sqi = *(const v8f*)(SQ + itile * 16 + 8 * hi);               // rows M
    int  ir  = itile * 16 + m;
    // f32 16x16x4 A layout: lanes0-15 hold K{0,1}, lanes16-31 K{2,3}
    v2f A0 = *(const v2f*)(F + ir * 8 + 2 * hi);       // K = 0..3
    v2f A1 = *(const v2f*)(F + ir * 8 + 4 + 2 * hi);   // K = 4..7

    float acc = 0.0f;
    for (int jt = wave; jt < NTILE; jt += 8) {
        int jr = jt * 16 + m;
        v2f B0 = *(const v2f*)(F + jr * 8 + 2 * hi);
        v2f B1 = *(const v2f*)(F + jr * 8 + 4 + 2 * hi);
        v16h sb  = *(const v16h*)(SB + jt * 512 + lane * 16);
        float sqj = SQ[jr];

        // Gram tile G = F_i . F_j^T  (f32, K=8 via two K=4 steps)
        v8f g = {};
        g = __builtin_amdgcn_wmma_f32_16x16x4_f32(false, A0, false, B0,
                                                  (short)0, g, false, false);
        g = __builtin_amdgcn_wmma_f32_16x16x4_f32(false, A1, false, B1,
                                                  (short)0, g, false, false);
        // M tile = (gate.segroi)^T . segroi over 21 channels (f16, K=32)
        v8f c = {};
        c = __builtin_amdgcn_wmma_f32_16x16x32_f16(false, sa, false, sb,
                                                   (short)0, c, false, false);

        // epilogue: acc += exp(-0.5*max(d2,0)) * M   (C/D layout: row v+8*hi)
        #pragma unroll
        for (int v = 0; v < 8; ++v) {
            float d2 = fmaxf(sqi[v] + sqj - 2.0f * g[v], 0.0f);
            acc = fmaf(__expf(-0.5f * d2), c[v], acc);
        }
    }

    // wave32 reduction + single atomic per wave
    #pragma unroll
    for (int off = 16; off > 0; off >>= 1)
        acc += __shfl_xor(acc, off, 32);
    if (lane == 0)
        atomicAdd(out, -2.5e-8f * acc);   // -(WEIGHT/N) = -(1e-7/4)
}

// ---------------------------------------------------------------------------
extern "C" void kernel_launch(void* const* d_in, const int* in_sizes, int n_in,
                              void* d_out, int out_size, void* d_ws, size_t ws_size,
                              hipStream_t stream)
{
    const float* images = (const float*)d_in[0];   // [4,3,128,128] f32
    const float* segs   = (const float*)d_in[1];   // [4,21,128,128] f32
    const float* rois   = (const float*)d_in[2];   // [4,128,128] f32
    const int*   labels = (const int*)d_in[3];     // [4,1,128,128] i32
    float* out = (float*)d_out;                    // [1] f32
    unsigned char* ws = (unsigned char*)d_ws;      // needs ~2.6 MB

    (void)in_sizes; (void)n_in; (void)out_size; (void)ws_size;

    dense_energy_prep<<<dim3((NIMG * PPIX) / 256), dim3(256), 0, stream>>>(
        images, segs, rois, labels, out, ws);
    dense_energy_main<<<dim3(NTILE, NIMG), dim3(256), 0, stream>>>(ws, out);
}
